// Mamba2_47244640256106
// MI455X (gfx1250) — matrix operand order repaired and checked by reading, not compiled, not run on hardware
//
#include <hip/hip_runtime.h>
#include <hip/hip_bf16.h>

// ---------------- model constants ----------------
#define BATCH   2
#define SEQ     2048
#define HIDDEN  2048
#define NHEADS  64
#define HDIM    64
#define STATE   128
#define CHUNK   256
#define NCHUNK  8            // SEQ / CHUNK
#define INTER   4096
#define CONVDIM 4352
#define PROJ    8512
#define CONVK   4

typedef __attribute__((ext_vector_type(16))) __bf16 v16bf;
typedef __attribute__((ext_vector_type(8)))  float  v8f;

// ---------------- TDM (Tensor Data Mover) support ----------------
typedef unsigned int tdm_v4u __attribute__((ext_vector_type(4)));
typedef int          tdm_v4i __attribute__((ext_vector_type(4)));
typedef int          tdm_v8i __attribute__((ext_vector_type(8)));

#if defined(__gfx1250__) && __has_builtin(__builtin_amdgcn_tensor_load_to_lds)
#define USE_TDM 1
#endif

// ---------------- async global->LDS support ----------------
#if defined(__gfx1250__) && __has_builtin(__builtin_amdgcn_global_load_async_to_lds_b128)
#define USE_ASYNC 1
typedef int async_v4i __attribute__((vector_size(16)));
typedef __attribute__((address_space(1))) async_v4i* async_gptr;
typedef __attribute__((address_space(3))) async_v4i* async_lptr;
#endif

// 16-byte global->LDS copy: async DMA when available (ASYNCcnt), else via VGPRs.
__device__ __forceinline__ void cp16_g2l(__bf16* lds, const __bf16* g) {
#ifdef USE_ASYNC
    __builtin_amdgcn_global_load_async_to_lds_b128((async_gptr)g, (async_lptr)lds, 0, 0);
#else
    *(uint4*)lds = *(const uint4*)g;
#endif
}

__device__ __forceinline__ void async_wait() {
#ifdef USE_ASYNC
#if __has_builtin(__builtin_amdgcn_s_wait_asynccnt)
    __builtin_amdgcn_s_wait_asynccnt(0);
#else
    asm volatile("s_wait_asynccnt 0x0" ::: "memory");
#endif
#endif
}

// Generic LDS pointer low 32 bits == LDS byte offset (flat aperture rule).
__device__ __forceinline__ unsigned lds_byte_off(const void* p) {
    return (unsigned)(unsigned long long)p;
}

// 2D tile load: rows x width (bf16 elements), global row stride = stride_elems.
__device__ __forceinline__ void tdm_load_2d(unsigned lds_addr, const void* gptr,
                                            unsigned width, unsigned rows,
                                            unsigned stride_elems) {
#ifdef USE_TDM
    unsigned long long ga = (unsigned long long)gptr;
    tdm_v4u g0 = { 1u, lds_addr, (unsigned)ga,
                   (unsigned)((ga >> 32) & 0x01FFFFFFull) | (2u << 30) };
    tdm_v8i g1 = { (int)0x00010000,                                  // data_size=1 (2B)
                   (int)((width & 0xffffu) << 16),                   // tensor_dim0 lo
                   (int)(((width >> 16) & 0xffffu) | ((rows & 0xffffu) << 16)),
                   (int)(((rows >> 16) & 0xffffu) | ((width & 0xffffu) << 16)), // tile_dim0
                   (int)(rows & 0xffffu),                            // tile_dim1
                   (int)stride_elems,                                // dim0_stride lo
                   0, 0 };
    tdm_v4i z4 = (tdm_v4i)0;
#if __clang_major__ >= 23
    tdm_v8i z8 = (tdm_v8i)0;
    __builtin_amdgcn_tensor_load_to_lds(g0, g1, z4, z4, z8, 0);
#else
    __builtin_amdgcn_tensor_load_to_lds(g0, g1, z4, z4, 0);
#endif
#endif
}

__device__ __forceinline__ void tdm_wait() {
#ifdef USE_TDM
#if __has_builtin(__builtin_amdgcn_s_wait_tensorcnt)
    __builtin_amdgcn_s_wait_tensorcnt(0);
#else
    asm volatile("s_wait_tensorcnt 0x0" ::: "memory");
#endif
#endif
}

// ---------------- WMMA helpers (gfx1250, wave32) ----------------
__device__ __forceinline__ v8f wmma_bf16(v16bf a, v16bf b, v8f c) {
    return __builtin_amdgcn_wmma_f32_16x16x32_bf16(false, a, false, b, (short)0, c,
                                                   false, false);
}

// A fragment, 16x32 bf16, source row-major [rows][ld] (contiguous reads).
__device__ __forceinline__ v16bf ldA(const __bf16* p, int row0, int k0, int ld, int lane) {
    int r = lane & 15, hh = lane >> 4;
    const __bf16* base = p + (size_t)(row0 + r) * ld + k0 + 8 * hh;
    v16bf f;
#pragma unroll
    for (int i = 0; i < 8; ++i) f[i] = base[i];
#pragma unroll
    for (int i = 0; i < 8; ++i) f[8 + i] = base[16 + i];
    return f;
}

// B fragment, 32x16 bf16, source row-major [K][ld]; lane%16 = column, table k order.
__device__ __forceinline__ v16bf ldB(const __bf16* p, int k0, int col0, int ld, int lane) {
    int c = lane & 15, hh = lane >> 4;
    const __bf16* base = p + (size_t)(k0 + 8 * hh) * ld + col0 + c;
    v16bf f;
#pragma unroll
    for (int i = 0; i < 8; ++i) f[i] = base[(size_t)i * ld];
#pragma unroll
    for (int i = 0; i < 8; ++i) f[8 + i] = base[(size_t)(16 + i) * ld];
    return f;
}

union U16x8 { uint4 u; __bf16 h[8]; };

// ---------------- Kernel: fp32 -> bf16 staging ----------------
__global__ void __launch_bounds__(256) cvt_bf16(const float* __restrict__ in,
                                                __bf16* __restrict__ out, int n) {
    int i4 = (blockIdx.x * 256 + threadIdx.x) * 4;
    if (i4 >= n) return;
    float4 v = *(const float4*)(in + i4);
    out[i4 + 0] = (__bf16)v.x;
    out[i4 + 1] = (__bf16)v.y;
    out[i4 + 2] = (__bf16)v.z;
    out[i4 + 3] = (__bf16)v.w;
}

// ---------------- GEMM: C[M,N] = A[M,K](bf16) * W[N,K](bf16)^T + bias ----------------
// 256 threads / 8 waves, block tile 256x64, wave tile 64x32, BK=32.
// A tiles arrive via async global->LDS DMA; W tile is transposed through VGPRs.
__global__ void __launch_bounds__(256) gemm_at_wt(const __bf16* __restrict__ A,
                                                  const __bf16* __restrict__ W,
                                                  const float* __restrict__ bias,
                                                  float* __restrict__ C,
                                                  int M, int N, int K) {
    __shared__ __bf16 As[256 * 40];     // 16B-aligned row stride
    __shared__ __bf16 Bs[32 * 72];      // [k][n], transposed W tile
    int tid = threadIdx.x, lane = tid & 31, wave = tid >> 5;
    int m0 = blockIdx.x * 256, n0 = blockIdx.y * 64;
    int wm = wave >> 1, wn = wave & 1;

    v8f acc[4][2] = {};

    for (int kt = 0; kt < K; kt += 32) {
#pragma unroll
        for (int i = 0; i < 4; ++i) {    // A tile 256x32 bf16: 16B async copies
            int e = tid + i * 256;
            int r = e >> 2, cq8 = (e & 3) * 8;
            cp16_g2l(As + r * 40 + cq8, A + (size_t)(m0 + r) * K + kt + cq8);
        }
        {                                // W tile 64x32 bf16, store transposed
            int r = tid >> 2, kq8 = (tid & 3) * 8;
            U16x8 u;
            u.u = *(const uint4*)(W + (size_t)(n0 + r) * K + kt + kq8);
#pragma unroll
            for (int j = 0; j < 8; ++j) Bs[(kq8 + j) * 72 + r] = u.h[j];
        }
        if (kt + 32 < K) {               // prefetch next tiles (global_prefetch_b8)
            __builtin_prefetch(A + (size_t)(m0 + tid) * K + kt + 32, 0, 1);
            __builtin_prefetch(W + (size_t)(n0 + (tid & 63)) * K + kt + 32, 0, 1);
        }
        async_wait();
        __syncthreads();

        v16bf b0 = ldB(Bs, 0, wn * 32, 72, lane);
        v16bf b1 = ldB(Bs, 0, wn * 32 + 16, 72, lane);
#pragma unroll
        for (int mi = 0; mi < 4; ++mi) {
            v16bf a = ldA(As, wm * 64 + mi * 16, 0, 40, lane);
            acc[mi][0] = wmma_bf16(a, b0, acc[mi][0]);
            acc[mi][1] = wmma_bf16(a, b1, acc[mi][1]);
        }
        __syncthreads();
    }

    int hh = lane >> 4, nn = lane & 15;
#pragma unroll
    for (int mi = 0; mi < 4; ++mi)
#pragma unroll
        for (int ni = 0; ni < 2; ++ni) {
            int coln = n0 + wn * 32 + ni * 16 + nn;
            float bv = bias[coln];
#pragma unroll
            for (int v = 0; v < 8; ++v) {
                int row = m0 + wm * 64 + mi * 16 + hh * 8 + v;
                C[(size_t)row * N + coln] = acc[mi][ni][v] + bv;
            }
        }
}

// ---------------- causal depthwise conv (K=4) + SiLU -> bf16 ----------------
__global__ void __launch_bounds__(256) conv_silu(const float* __restrict__ proj,
                                                 const float* __restrict__ cw,
                                                 __bf16* __restrict__ xh) {
    size_t idx = (size_t)blockIdx.x * 256 + threadIdx.x;
    if (idx >= (size_t)BATCH * SEQ * CONVDIM) return;
    int ch = (int)(idx % CONVDIM);
    int t  = (int)((idx / CONVDIM) % SEQ);
    int b  = (int)(idx / ((size_t)CONVDIM * SEQ));
    float acc = 0.f;
#pragma unroll
    for (int j = 0; j < CONVK; ++j) {
        int tt = t - (CONVK - 1) + j;
        if (tt >= 0)
            acc += proj[((size_t)(b * SEQ + tt)) * PROJ + INTER + ch] * cw[ch * CONVK + j];
    }
    xh[idx] = (__bf16)(acc / (1.f + __expf(-acc)));    // SiLU
}

// ---------------- dt = softplus(dt+bias), Acum = cumsum(A*dt) per chunk ----------------
__global__ void __launch_bounds__(256) dt_prep(const float* __restrict__ proj,
                                               const float* __restrict__ dt_bias,
                                               const float* __restrict__ A_log,
                                               float* __restrict__ dtbuf,
                                               float* __restrict__ acum) {
    int blk = blockIdx.x;                 // ((b*H + h)*NCHUNK + c)
    int h = (blk >> 3) & (NHEADS - 1);
    int b = blk >> 9;
    int c = blk & (NCHUNK - 1);
    int s = threadIdx.x;
    size_t tok = (size_t)b * SEQ + c * CHUNK + s;
    float raw = proj[tok * PROJ + INTER + CONVDIM + h] + dt_bias[h];
    float dtv = raw > 20.f ? raw : log1pf(__expf(raw));
    float Aval = -__expf(A_log[h]);
    __shared__ float sh[CHUNK];
    sh[s] = Aval * dtv;
    __syncthreads();
    for (int off = 1; off < CHUNK; off <<= 1) {
        float v = (s >= off) ? sh[s - off] : 0.f;
        __syncthreads();
        sh[s] += v;
        __syncthreads();
    }
    dtbuf[(size_t)blk * CHUNK + s] = dtv;
    acum[(size_t)blk * CHUNK + s]  = sh[s];
}

// ---------------- per-(b,c,h) intra-chunk Y_diag + chunk states ----------------
// ~289KB dynamic LDS (CDNA5 WGP: 320KB). B/C tiles arrive via TDM.
__global__ void __launch_bounds__(256) ssm_chunk_a(const __bf16* __restrict__ xh,
                                                   const float* __restrict__ dtbuf,
                                                   const float* __restrict__ acum,
                                                   float* __restrict__ ybuf,
                                                   float* __restrict__ states) {
    extern __shared__ char smem[];
    __bf16* Bsh   = (__bf16*)smem;          // [256][128]
    __bf16* Csh   = Bsh + 256 * 128;        // [256][128]
    __bf16* BshT  = Csh + 256 * 128;        // [128][264]  B^T for G stage
    __bf16* Xsh   = BshT + 128 * 264;       // [256][64]   x*dt
    __bf16* XdecT = Xsh + 256 * 64;         // [64][264]   (x*dt*decay)^T
    __bf16* Gm    = XdecT + 64 * 264;       // 8 waves * [32][40]
    float*  dts   = (float*)(Gm + 8 * 32 * 40);
    float*  acs   = dts + 256;
    float*  dec   = acs + 256;

    int blk = blockIdx.x;
    int h = blk & 63, c = (blk >> 6) & 7, b = blk >> 9;
    int tid = threadIdx.x, lane = tid & 31, wave = tid >> 5;
    size_t tok0 = (size_t)b * SEQ + c * CHUNK;
    size_t dtbase = (((size_t)b * NHEADS + h) * NCHUNK + c) * CHUNK;

    dts[tid] = dtbuf[dtbase + tid];
    acs[tid] = acum[dtbase + tid];
#ifdef USE_TDM
    if (tid < 32) {       // one wave issues both tile DMAs (EXEC ignored by TDM)
        tdm_load_2d(lds_byte_off(Bsh), xh + tok0 * CONVDIM + INTER, 128, 256, CONVDIM);
        tdm_load_2d(lds_byte_off(Csh), xh + tok0 * CONVDIM + INTER + STATE, 128, 256, CONVDIM);
    }
#endif
    __syncthreads();
    float Alast = acs[CHUNK - 1];
    dec[tid] = __expf(Alast - acs[tid]);
    __syncthreads();
    for (int i = tid; i < 256 * 64; i += 256) {      // x*dt and transposed decayed copy
        int s = i >> 6, p = i & 63;
        float xd = (float)xh[(tok0 + s) * CONVDIM + h * HDIM + p] * dts[s];
        Xsh[i] = (__bf16)xd;
        XdecT[p * 264 + s] = (__bf16)(xd * dec[s]);
    }
#ifndef USE_TDM
    for (int i = tid; i < 256 * 128; i += 256) {
        int s = i >> 7, n = i & 127;
        const __bf16* row = xh + (tok0 + s) * CONVDIM;
        Bsh[i] = row[INTER + n];
        Csh[i] = row[INTER + STATE + n];
    }
#endif
#ifdef USE_TDM
    if (tid < 32) tdm_wait();
#endif
    __syncthreads();
    for (int i = tid; i < 256 * 16; i += 256) {      // build B^T (LDS->LDS, vector reads)
        int s = i >> 4, n8 = (i & 15) * 8;
        U16x8 u;
        u.u = *(const uint4*)(Bsh + s * 128 + n8);
#pragma unroll
        for (int j = 0; j < 8; ++j) BshT[(n8 + j) * 264 + s] = u.h[j];
    }
    __syncthreads();

    // ---- stage 1: Y_diag, one 32-row t-block per wave ----
    int tb = wave * 32;
    int hh = lane >> 4, nn = lane & 15;
    __bf16* gm = Gm + wave * 32 * 40;
    v8f yacc[2][4] = {};
    for (int sb = 0; sb <= wave; ++sb) {
        int sbase = sb * 32;
        v8f g[2][2] = {};
#pragma unroll
        for (int kk = 0; kk < STATE; kk += 32) {     // G = C * B^T over n
            v16bf ca0 = ldA(Csh, tb, kk, 128, lane);
            v16bf ca1 = ldA(Csh, tb + 16, kk, 128, lane);
            v16bf bb0 = ldB(BshT, kk, sbase, 264, lane);
            v16bf bb1 = ldB(BshT, kk, sbase + 16, 264, lane);
            g[0][0] = wmma_bf16(ca0, bb0, g[0][0]);
            g[0][1] = wmma_bf16(ca0, bb1, g[0][1]);
            g[1][0] = wmma_bf16(ca1, bb0, g[1][0]);
            g[1][1] = wmma_bf16(ca1, bb1, g[1][1]);
        }
        // decay mask L[t,s] = exp(Acum[t]-Acum[s]) for s<=t else 0, restage bf16
#pragma unroll
        for (int mi = 0; mi < 2; ++mi)
#pragma unroll
            for (int ni = 0; ni < 2; ++ni)
#pragma unroll
                for (int v = 0; v < 8; ++v) {
                    int tr = mi * 16 + hh * 8 + v;
                    int sc = ni * 16 + nn;
                    int ti = tb + tr, si = sbase + sc;
                    float val = (si <= ti) ? g[mi][ni][v] * __expf(acs[ti] - acs[si]) : 0.f;
                    gm[tr * 40 + sc] = (__bf16)val;   // wave-private
                }
        v16bf ga0 = ldA(gm, 0, 0, 40, lane);
        v16bf ga1 = ldA(gm, 16, 0, 40, lane);
#pragma unroll
        for (int pb = 0; pb < 4; ++pb) {             // Y += Gm * (x*dt)
            v16bf xb = ldB(Xsh, sbase, pb * 16, 64, lane);
            yacc[0][pb] = wmma_bf16(ga0, xb, yacc[0][pb]);
            yacc[1][pb] = wmma_bf16(ga1, xb, yacc[1][pb]);
        }
    }
#pragma unroll
    for (int mi = 0; mi < 2; ++mi)
#pragma unroll
        for (int pb = 0; pb < 4; ++pb)
#pragma unroll
            for (int v = 0; v < 8; ++v) {
                int t = tb + mi * 16 + hh * 8 + v;
                int p = pb * 16 + nn;
                ybuf[(tok0 + t) * INTER + h * HDIM + p] = yacc[mi][pb][v];
            }

    // ---- stage 2: states[p,n] = sum_s (x*dt*decay)[s,p] * B[s,n] ----
    int wp = wave >> 2, wn2 = wave & 3;
    v8f sacc[2][2] = {};
    for (int sb = 0; sb < CHUNK; sb += 32) {
        v16bf xa0 = ldA(XdecT, wp * 32, sb, 264, lane);
        v16bf xa1 = ldA(XdecT, wp * 32 + 16, sb, 264, lane);
        v16bf bb0 = ldB(Bsh, sb, wn2 * 32, 128, lane);
        v16bf bb1 = ldB(Bsh, sb, wn2 * 32 + 16, 128, lane);
        sacc[0][0] = wmma_bf16(xa0, bb0, sacc[0][0]);
        sacc[0][1] = wmma_bf16(xa0, bb1, sacc[0][1]);
        sacc[1][0] = wmma_bf16(xa1, bb0, sacc[1][0]);
        sacc[1][1] = wmma_bf16(xa1, bb1, sacc[1][1]);
    }
    size_t stbase = (((size_t)b * NCHUNK + c) * NHEADS + h) * (HDIM * STATE);
#pragma unroll
    for (int mi = 0; mi < 2; ++mi)
#pragma unroll
        for (int ni = 0; ni < 2; ++ni)
#pragma unroll
            for (int v = 0; v < 8; ++v) {
                int p = wp * 32 + mi * 16 + hh * 8 + v;
                int n = wn2 * 32 + ni * 16 + nn;
                states[stbase + p * STATE + n] = sacc[mi][ni][v];
            }
}

// ---------------- inter-chunk recurrence ----------------
__global__ void __launch_bounds__(256) recur(const float* __restrict__ states,
                                             const float* __restrict__ acum,
                                             float* __restrict__ prev) {
    int b = blockIdx.x >> 6, h = blockIdx.x & 63;
    int tid = threadIdx.x;
    float pv[32];
#pragma unroll
    for (int i = 0; i < 32; ++i) pv[i] = 0.f;
    for (int c = 0; c < NCHUNK; ++c) {
        size_t base = (((size_t)b * NCHUNK + c) * NHEADS + h) * (HDIM * STATE);
#pragma unroll
        for (int i = 0; i < 32; ++i) prev[base + i * 256 + tid] = pv[i];
        float al = __expf(acum[(((size_t)b * NHEADS + h) * NCHUNK + c) * CHUNK + CHUNK - 1]);
#pragma unroll
        for (int i = 0; i < 32; ++i) pv[i] = al * pv[i] + states[base + i * 256 + tid];
    }
}

// ---------------- Y_off = exp(Acum) * (C * prev^T) + combine + D skip ----------------
__global__ void __launch_bounds__(256) ssm_chunk_b(const __bf16* __restrict__ xh,
                                                   const float* __restrict__ acum,
                                                   const float* __restrict__ prev,
                                                   const float* __restrict__ Dvec,
                                                   float* __restrict__ ybuf) {
    extern __shared__ char smem[];
    __bf16* Csh = (__bf16*)smem;            // [256][128]
    __bf16* PvT = Csh + 256 * 128;          // [128][72]  prev^T (n-major)
    float*  acs = (float*)(PvT + 128 * 72);

    int blk = blockIdx.x;
    int h = blk & 63, c = (blk >> 6) & 7, b = blk >> 9;
    int tid = threadIdx.x, lane = tid & 31, wave = tid >> 5;
    size_t tok0 = (size_t)b * SEQ + c * CHUNK;
    size_t pvbase = (((size_t)b * NCHUNK + c) * NHEADS + h) * (HDIM * STATE);

#ifdef USE_TDM
    if (tid < 32)
        tdm_load_2d(lds_byte_off(Csh), xh + tok0 * CONVDIM + INTER + STATE, 128, 256, CONVDIM);
#else
    for (int i = tid; i < 256 * 128; i += 256) {
        int s = i >> 7, n = i & 127;
        Csh[i] = xh[(tok0 + s) * CONVDIM + INTER + STATE + n];
    }
#endif
    for (int i = tid; i < 128 * 64; i += 256) {
        int n = i >> 6, p = i & 63;
        PvT[n * 72 + p] = (__bf16)prev[pvbase + p * STATE + n];
    }
    acs[tid] = acum[(((size_t)b * NHEADS + h) * NCHUNK + c) * CHUNK + tid];
#ifdef USE_TDM
    if (tid < 32) tdm_wait();
#endif
    __syncthreads();

    int tb = wave * 32;
    int hh = lane >> 4, nn = lane & 15;
    v8f acc[2][4] = {};
#pragma unroll
    for (int kk = 0; kk < STATE; kk += 32) {
        v16bf ca0 = ldA(Csh, tb, kk, 128, lane);
        v16bf ca1 = ldA(Csh, tb + 16, kk, 128, lane);
#pragma unroll
        for (int pb = 0; pb < 4; ++pb) {
            v16bf bp = ldB(PvT, kk, pb * 16, 72, lane);
            acc[0][pb] = wmma_bf16(ca0, bp, acc[0][pb]);
            acc[1][pb] = wmma_bf16(ca1, bp, acc[1][pb]);
        }
    }
    float Dh = Dvec[h];
#pragma unroll
    for (int mi = 0; mi < 2; ++mi)
#pragma unroll
        for (int pb = 0; pb < 4; ++pb)
#pragma unroll
            for (int v = 0; v < 8; ++v) {
                int t = tb + mi * 16 + hh * 8 + v;
                int p = pb * 16 + nn;
                size_t yi = (tok0 + t) * INTER + h * HDIM + p;
                float xv = (float)xh[(tok0 + t) * CONVDIM + h * HDIM + p];
                ybuf[yi] = ybuf[yi] + acc[mi][pb][v] * __expf(acs[t]) + Dh * xv;
            }
}

// ---------------- gated RMSNorm -> bf16 activations for out_proj ----------------
__global__ void __launch_bounds__(256) gated_norm(const float* __restrict__ proj,
                                                  const float* __restrict__ norm_w,
                                                  const float* __restrict__ ybuf,
                                                  __bf16* __restrict__ gh) {
    int tok = blockIdx.x;
    int tid = threadIdx.x;
    __shared__ float red[256];
    float gl[16];
    float ss = 0.f;
#pragma unroll
    for (int i = 0; i < 16; ++i) {
        int e = tid + i * 256;
        float gate = proj[(size_t)tok * PROJ + e];
        float gv = ybuf[(size_t)tok * INTER + e] * (gate / (1.f + __expf(-gate)));
        gl[i] = gv;
        ss += gv * gv;
    }
    red[tid] = ss;
    __syncthreads();
    for (int off = 128; off > 0; off >>= 1) {
        if (tid < off) red[tid] += red[tid + off];
        __syncthreads();
    }
    float scale = rsqrtf(red[0] / (float)INTER + 1e-5f);
#pragma unroll
    for (int i = 0; i < 16; ++i) {
        int e = tid + i * 256;
        gh[(size_t)tok * INTER + e] = (__bf16)(gl[i] * scale * norm_w[e]);
    }
}

// ---------------- launch ----------------
extern "C" void kernel_launch(void* const* d_in, const int* in_sizes, int n_in,
                              void* d_out, int out_size, void* d_ws, size_t ws_size,
                              hipStream_t stream) {
    const float* hidden  = (const float*)d_in[0];
    const float* in_w    = (const float*)d_in[1];
    const float* in_b    = (const float*)d_in[2];
    const float* conv_w  = (const float*)d_in[3];
    const float* dt_bias = (const float*)d_in[4];
    const float* A_log   = (const float*)d_in[5];
    const float* Dvec    = (const float*)d_in[6];
    const float* norm_w  = (const float*)d_in[7];
    const float* out_w   = (const float*)d_in[8];
    const float* out_b   = (const float*)d_in[9];
    float* out = (float*)d_out;

    const int TOK = BATCH * SEQ;                 // 4096
    float* ws = (float*)d_ws;
    float* proj   = ws;                                      // TOK*PROJ f32
    float* ybuf   = proj   + (size_t)TOK * PROJ;             // TOK*INTER f32
    float* states = ybuf   + (size_t)TOK * INTER;
    float* prevb  = states + (size_t)BATCH * NCHUNK * NHEADS * HDIM * STATE;
    float* dtbuf  = prevb  + (size_t)BATCH * NCHUNK * NHEADS * HDIM * STATE;
    float* acum   = dtbuf  + (size_t)BATCH * NHEADS * SEQ;
    float* fend   = acum   + (size_t)BATCH * NHEADS * SEQ;
    __bf16* xh       = (__bf16*)fend;                        // TOK*CONVDIM bf16
    __bf16* hidden_h = xh + (size_t)TOK * CONVDIM;           // TOK*HIDDEN
    __bf16* in_w_h   = hidden_h + (size_t)TOK * HIDDEN;      // PROJ*HIDDEN
    __bf16* out_w_h  = in_w_h + (size_t)PROJ * HIDDEN;       // HIDDEN*INTER
    __bf16* g_h      = out_w_h + (size_t)HIDDEN * INTER;     // TOK*INTER

    dim3 blk(256);
    // 0) bf16 staging of GEMM operands
    cvt_bf16<<<(TOK * HIDDEN / 4 + 255) / 256, blk, 0, stream>>>(hidden, hidden_h, TOK * HIDDEN);
    cvt_bf16<<<(PROJ * HIDDEN / 4 + 255) / 256, blk, 0, stream>>>(in_w, in_w_h, PROJ * HIDDEN);
    cvt_bf16<<<(HIDDEN * INTER / 4 + 255) / 256, blk, 0, stream>>>(out_w, out_w_h, HIDDEN * INTER);
    // 1) in_proj
    gemm_at_wt<<<dim3(TOK / 256, PROJ / 64), blk, 0, stream>>>(
        hidden_h, in_w_h, in_b, proj, TOK, PROJ, HIDDEN);
    // 2) causal conv + SiLU (bf16 out)
    conv_silu<<<(unsigned)(((size_t)TOK * CONVDIM + 255) / 256), blk, 0, stream>>>(
        proj, conv_w, xh);
    // 3) dt softplus + A*dt chunk cumsum
    dt_prep<<<BATCH * NHEADS * NCHUNK, blk, 0, stream>>>(proj, dt_bias, A_log, dtbuf, acum);
    // 4) intra-chunk Y_diag + chunk states (TDM tile loads, ~289KB LDS)
    size_t shA = (size_t)(256 * 128 * 2 * 2 + 128 * 264 * 2 + 256 * 64 * 2 + 64 * 264 * 2 +
                          8 * 32 * 40 * 2 + 3 * 256 * 4);
    ssm_chunk_a<<<BATCH * NCHUNK * NHEADS, blk, shA, stream>>>(xh, dtbuf, acum, ybuf, states);
    // 5) inter-chunk recurrence
    recur<<<BATCH * NHEADS, blk, 0, stream>>>(states, acum, prevb);
    // 6) Y_off + combine + D skip
    size_t shB = (size_t)(256 * 128 * 2 + 128 * 72 * 2 + 256 * 4);
    ssm_chunk_b<<<BATCH * NCHUNK * NHEADS, blk, shB, stream>>>(xh, acum, prevb, Dvec, ybuf);
    // 7) gated RMSNorm -> bf16
    gated_norm<<<TOK, blk, 0, stream>>>(proj, norm_w, ybuf, g_h);
    // 8) out_proj
    gemm_at_wt<<<dim3(TOK / 256, HIDDEN / 64), blk, 0, stream>>>(
        g_h, out_w_h, out_b, out, TOK, HIDDEN, INTER);
}
